// SlotAttentionPosEmbed_42288247996586
// MI455X (gfx1250) — matrix operand order repaired and checked by reading, not compiled, not run on hardware
//
#include <hip/hip_runtime.h>
#include <hip/hip_bf16.h>
#include <math.h>

typedef _Float16 v16h __attribute__((ext_vector_type(16)));
typedef _Float16 v8h  __attribute__((ext_vector_type(8)));
typedef _Float16 v4h  __attribute__((ext_vector_type(4)));
typedef float    v8f  __attribute__((ext_vector_type(8)));

#define BB 64
#define HH 128
#define WW 128
#define HWSZ 16384          // H*W
#define CC 64
#define DD 64
#define NPIX 1048576        // B*H*W
#define KCONV 1600          // 64*25

// ---------------------------------------------------------------------------
// Weight prep: f32 -> f16, transposed into B^T ([out][in]) layout for WMMA.
// conv weights [co][ci][ky][kx] are already [co][k] flat with k = ci*25+ky*5+kx.
// ---------------------------------------------------------------------------
__global__ void prep_weights_k(const float* __restrict__ c2w, const float* __restrict__ c3w,
                               const float* __restrict__ m1,  const float* __restrict__ m2,
                               const float* __restrict__ kw,  const float* __restrict__ vw,
                               _Float16* __restrict__ w2T, _Float16* __restrict__ w3T,
                               _Float16* __restrict__ m1T, _Float16* __restrict__ m2T,
                               _Float16* __restrict__ kT,  _Float16* __restrict__ vT) {
    int i = blockIdx.x * 256 + threadIdx.x;
    if (i < 64 * KCONV) {
        w2T[i] = (_Float16)c2w[i];
        w3T[i] = (_Float16)c3w[i];
    }
    if (i < 64 * 64) {
        int r = i >> 6, c = i & 63;        // w[r][c] -> wT[c][r]
        m1T[c * 64 + r] = (_Float16)m1[i];
        m2T[c * 64 + r] = (_Float16)m2[i];
        kT [c * 64 + r] = (_Float16)kw[i];
        vT [c * 64 + r] = (_Float16)vw[i];
    }
}

// ---------------------------------------------------------------------------
// conv1: 1 -> 64 channels, 5x5, pad 2, relu. Direct f32 (only ~3 GFLOP).
// Output f16 NCHW.
// ---------------------------------------------------------------------------
__global__ __launch_bounds__(256) void conv1_k(const float* __restrict__ x,
                                               const float* __restrict__ w,
                                               const float* __restrict__ bias,
                                               _Float16* __restrict__ out) {
    __shared__ float wl[64 * 25];
    __shared__ float bl[64];
    for (int i = threadIdx.x; i < 64 * 25; i += 256) wl[i] = w[i];
    if (threadIdx.x < 64) bl[threadIdx.x] = bias[threadIdx.x];
    __syncthreads();

    int p  = blockIdx.x * 256 + threadIdx.x;
    int b  = p >> 14;
    int yx = p & 16383;
    int y  = yx >> 7;
    int x0 = yx & 127;

    float v[25];
#pragma unroll
    for (int ky = 0; ky < 5; ky++)
#pragma unroll
        for (int kx = 0; kx < 5; kx++) {
            int yy = y + ky - 2, xx = x0 + kx - 2;
            v[ky * 5 + kx] = (yy >= 0 && yy < HH && xx >= 0 && xx < WW)
                                 ? x[(b << 14) + (yy << 7) + xx] : 0.f;
        }
    for (int co = 0; co < 64; co++) {
        float a = bl[co];
#pragma unroll
        for (int t = 0; t < 25; t++) a += v[t] * wl[co * 25 + t];
        a = fmaxf(a, 0.f);
        out[((size_t)((b << 6) + co) << 14) + yx] = (_Float16)a;
    }
}

// ---------------------------------------------------------------------------
// conv2/conv3 as implicit GEMM with WMMA.
// Block = 128 threads (4 waves). Tile = 16 consecutive pixels x 64 Cout.
// A staged as a 16x64 f16 slab in LDS per round (2 WMMA K-steps / barrier
// pair); B read from [Cout][1600] f16 global (hot in L2).
// ---------------------------------------------------------------------------
__global__ __launch_bounds__(128) void conv_wmma_k(const _Float16* __restrict__ xin,
                                                   const _Float16* __restrict__ wT,
                                                   const float* __restrict__ bias,
                                                   _Float16* __restrict__ out,
                                                   int do_relu) {
    __shared__ _Float16 Atile[16 * 72];      // row stride 72 halfs (bank friendly)

    int tile = blockIdx.x;                   // NPIX/16 tiles
    int b    = tile >> 10;                   // 1024 tiles per image
    int rem  = tile & 1023;
    int y    = rem >> 3;                     // 8 tiles per row
    int x0   = (rem & 7) << 4;

    int lane = threadIdx.x & 31;
    int wv   = threadIdx.x >> 5;
    int m_l  = lane & 15;
    int g0A  = (lane < 16) ? 0 : 16;         // A fragment K group base
    int g0B  = (lane < 16) ? 0 : 16;         // B fragment K base
    int co   = (wv << 4) + m_l;              // this lane's output channel (N)

    int fillm  = threadIdx.x >> 3;           // 0..15 : A row to fill
    int fillk0 = (threadIdx.x & 7) << 3;     // 0,8,...,56

    v8f acc = {};

    for (int kb = 0; kb < KCONV; kb += 64) {
        // ---- cooperative A fill (16x64, implicit im2col + zero padding) ----
        {
            v8h tv;
#pragma unroll
            for (int j = 0; j < 8; j++) {
                int k  = kb + fillk0 + j;
                int ci = k / 25;
                int r  = k - ci * 25;
                int ky = r / 5;
                int kx = r - ky * 5;
                int yy = y + ky - 2;
                int xx = x0 + fillm + kx - 2;
                tv[j] = (yy >= 0 && yy < HH && xx >= 0 && xx < WW)
                            ? xin[((size_t)((b << 6) + ci) << 14) + (yy << 7) + xx]
                            : (_Float16)0.f;
            }
            *(v8h*)&Atile[fillm * 72 + fillk0] = tv;
        }
        __syncthreads();

        // ---- two K=32 WMMA steps per staged slab ----
#pragma unroll
        for (int ks = 0; ks < 2; ks++) {
            int kof = ks << 5;
            v8h alo = *(const v8h*)&Atile[m_l * 72 + kof + g0A];
            v8h ahi = *(const v8h*)&Atile[m_l * 72 + kof + g0A + 8];
            const v8h* bp = (const v8h*)&wT[(size_t)co * KCONV + kb + kof + g0B];
            v8h b0 = bp[0];
            v8h b1 = bp[1];

            v16h a, bf;
#pragma unroll
            for (int i = 0; i < 8; i++) {
                a[i] = alo[i]; a[8 + i] = ahi[i];
                bf[i] = b0[i]; bf[8 + i] = b1[i];
            }
            acc = __builtin_amdgcn_wmma_f32_16x16x32_f16(false, a, false, bf,
                                                         (short)0, acc, false, false);
        }
        __syncthreads();
    }

    // ---- epilogue: bias (+relu), contiguous 8-half store per lane ----
    float bs   = bias[co];
    int   mb   = (lane >> 4) << 3;           // 0 or 8
    v8h   o;
#pragma unroll
    for (int r = 0; r < 8; r++) {
        float t = acc[r] + bs;
        if (do_relu) t = fmaxf(t, 0.f);
        o[r] = (_Float16)t;
    }
    *(v8h*)&out[((size_t)((b << 6) + co) << 14) + (y << 7) + x0 + mb] = o;
}

// ---------------------------------------------------------------------------
// pos-embed + LayerNorm: NCHW f16 conv3 output -> row-major [pixel][64] f16.
// ---------------------------------------------------------------------------
__global__ __launch_bounds__(256) void lnpos_k(const _Float16* __restrict__ x3,
                                               const float* __restrict__ pw,
                                               const float* __restrict__ pb,
                                               const float* __restrict__ g,
                                               const float* __restrict__ bta,
                                               _Float16* __restrict__ outln) {
    __shared__ _Float16 tile[256 * 68];
    int p  = blockIdx.x * 256 + threadIdx.x;
    int b  = p >> 14;
    int yx = p & 16383;
    int y  = yx >> 7;
    int xx = yx & 127;
    float gy = (float)y  * (1.f / 127.f);
    float gx = (float)xx * (1.f / 127.f);

    float s = 0.f, s2 = 0.f;
    for (int c = 0; c < 64; c++) {
        float v = (float)x3[((size_t)((b << 6) + c) << 14) + yx];
        v += gy * pw[c] + gx * pw[64 + c] + (1.f - gy) * pw[128 + c] +
             (1.f - gx) * pw[192 + c] + pb[c];
        tile[threadIdx.x * 68 + c] = (_Float16)v;
        s += v; s2 += v * v;
    }
    float mu  = s * (1.f / 64.f);
    float var = fmaxf(s2 * (1.f / 64.f) - mu * mu, 0.f);
    float rs  = rsqrtf(var + 1e-5f);

    for (int c0 = 0; c0 < 64; c0 += 8) {
        v8h o;
#pragma unroll
        for (int j = 0; j < 8; j++) {
            int c   = c0 + j;
            float v = (float)tile[threadIdx.x * 68 + c];
            o[j] = (_Float16)((v - mu) * rs * g[c] + bta[c]);
        }
        *(v8h*)&outln[(size_t)p * 64 + c0] = o;
    }
}

// ---------------------------------------------------------------------------
// Row GEMM (K=64) with WMMA: out[M][64] = act(in[M][64] @ W^T + bias).
// Block = 128 threads (4 waves), tile = 16 rows x 64 cols.
// ---------------------------------------------------------------------------
__global__ __launch_bounds__(128) void rowgemm_k(const _Float16* __restrict__ in,
                                                 const _Float16* __restrict__ wT,
                                                 const float* __restrict__ bias,
                                                 _Float16* __restrict__ out,
                                                 int do_relu) {
    __shared__ float ctile[16 * 65];
    int tile0 = blockIdx.x << 4;
    int lane  = threadIdx.x & 31;
    int wv    = threadIdx.x >> 5;
    int m_l   = lane & 15;
    int g0A   = (lane < 16) ? 0 : 16;
    int g0B   = (lane < 16) ? 0 : 16;
    int row   = tile0 + m_l;
    int co    = (wv << 4) + m_l;

    v8f acc = {};
#pragma unroll
    for (int kb = 0; kb < 64; kb += 32) {
        v8h alo = *(const v8h*)&in[(size_t)row * 64 + kb + g0A];
        v8h ahi = *(const v8h*)&in[(size_t)row * 64 + kb + g0A + 8];
        v8h b0  = *(const v8h*)&wT[(size_t)co * 64 + kb + g0B];
        v8h b1  = *(const v8h*)&wT[(size_t)co * 64 + kb + g0B + 8];
        v16h a, bf;
#pragma unroll
        for (int i = 0; i < 8; i++) {
            a[i] = alo[i]; a[8 + i] = ahi[i];
            bf[i] = b0[i]; bf[8 + i] = b1[i];
        }
        acc = __builtin_amdgcn_wmma_f32_16x16x32_f16(false, a, false, bf,
                                                     (short)0, acc, false, false);
    }
    float bs = bias[co];
    int   mb = (lane >> 4) << 3;
#pragma unroll
    for (int r = 0; r < 8; r++) {
        float t = acc[r] + bs;
        if (do_relu) t = fmaxf(t, 0.f);
        ctile[(mb + r) * 65 + co] = t;
    }
    __syncthreads();
    int r   = threadIdx.x >> 3;
    int seg = (threadIdx.x & 7) << 3;
    v8h o;
#pragma unroll
    for (int j = 0; j < 8; j++) o[j] = (_Float16)ctile[r * 65 + seg + j];
    *(v8h*)&out[(size_t)(tile0 + r) * 64 + seg] = o;
}

// ---------------------------------------------------------------------------
// Fused toK/toV GEMM. keys written in-place over enc; vals to other buffer.
// ---------------------------------------------------------------------------
__global__ __launch_bounds__(128) void kv_k(const _Float16* __restrict__ enc,
                                            const _Float16* __restrict__ kT,
                                            const _Float16* __restrict__ vT,
                                            const float* __restrict__ kb_,
                                            const float* __restrict__ vb_,
                                            _Float16* __restrict__ keys,
                                            _Float16* __restrict__ vals) {
    __shared__ float ck[16 * 65];
    __shared__ float cv[16 * 65];
    int tile0 = blockIdx.x << 4;
    int lane  = threadIdx.x & 31;
    int wv    = threadIdx.x >> 5;
    int m_l   = lane & 15;
    int g0A   = (lane < 16) ? 0 : 16;
    int g0B   = (lane < 16) ? 0 : 16;
    int row   = tile0 + m_l;
    int co    = (wv << 4) + m_l;

    v8f accK = {};
    v8f accV = {};
#pragma unroll
    for (int kb = 0; kb < 64; kb += 32) {
        v8h alo = *(const v8h*)&enc[(size_t)row * 64 + kb + g0A];
        v8h ahi = *(const v8h*)&enc[(size_t)row * 64 + kb + g0A + 8];
        v8h k0  = *(const v8h*)&kT[(size_t)co * 64 + kb + g0B];
        v8h k1  = *(const v8h*)&kT[(size_t)co * 64 + kb + g0B + 8];
        v8h v0  = *(const v8h*)&vT[(size_t)co * 64 + kb + g0B];
        v8h v1  = *(const v8h*)&vT[(size_t)co * 64 + kb + g0B + 8];
        v16h a, bk, bv;
#pragma unroll
        for (int i = 0; i < 8; i++) {
            a[i] = alo[i]; a[8 + i] = ahi[i];
            bk[i] = k0[i]; bk[8 + i] = k1[i];
            bv[i] = v0[i]; bv[8 + i] = v1[i];
        }
        accK = __builtin_amdgcn_wmma_f32_16x16x32_f16(false, a, false, bk,
                                                      (short)0, accK, false, false);
        accV = __builtin_amdgcn_wmma_f32_16x16x32_f16(false, a, false, bv,
                                                      (short)0, accV, false, false);
    }
    float bk_s = kb_[co], bv_s = vb_[co];
    int   mb   = (lane >> 4) << 3;
#pragma unroll
    for (int r = 0; r < 8; r++) {
        ck[(mb + r) * 65 + co] = accK[r] + bk_s;
        cv[(mb + r) * 65 + co] = accV[r] + bv_s;
    }
    __syncthreads();
    int r   = threadIdx.x >> 3;
    int seg = (threadIdx.x & 7) << 3;
    v8h ok, ov;
#pragma unroll
    for (int j = 0; j < 8; j++) {
        ok[j] = (_Float16)ck[r * 65 + seg + j];
        ov[j] = (_Float16)cv[r * 65 + seg + j];
    }
    *(v8h*)&keys[(size_t)(tile0 + r) * 64 + seg] = ok;
    *(v8h*)&vals[(size_t)(tile0 + r) * 64 + seg] = ov;
}

// ---------------------------------------------------------------------------
// Async global->LDS staging helper (CDNA5 GLOBAL_LOAD_ASYNC_TO_LDS_B128,
// GVS mode: SGPR64 base + per-lane signed 32-bit byte offset). ASYNCcnt
// tracked; caller must s_wait_asynccnt + barrier before consuming LDS.
// ---------------------------------------------------------------------------
__device__ __forceinline__ void async_b128_to_lds(_Float16* lds_dst,
                                                  const _Float16* gbase,
                                                  int byte_off) {
    unsigned lds = (unsigned)(size_t)lds_dst;      // low 32 bits = LDS offset
    asm volatile("global_load_async_to_lds_b128 %0, %1, %2"
                 :
                 : "v"(lds), "v"(byte_off), "s"(gbase)
                 : "memory");
}

__device__ __forceinline__ void wait_async0() {
    asm volatile("s_wait_asynccnt 0x0" ::: "memory");
}

// ---------------------------------------------------------------------------
// Slot attention: one 512-thread block per batch image runs all 3 iterations
// (LN -> attn -> GRU) plus the final att/wts output. f32 VALU (tiny FLOPs);
// keys/vals chunks staged with async-to-LDS DMA.
// ---------------------------------------------------------------------------
__global__ __launch_bounds__(512) void attn_k(const _Float16* __restrict__ keys,
                                              const _Float16* __restrict__ vals,
                                              const float* __restrict__ qinit,
                                              const float* __restrict__ qg,
                                              const float* __restrict__ qb,
                                              const float* __restrict__ wih,
                                              const float* __restrict__ whh,
                                              const float* __restrict__ bih,
                                              const float* __restrict__ bhh,
                                              float* __restrict__ dout) {
    __shared__ _Float16 kch[64 * 72];
    __shared__ _Float16 vch[64 * 72];
    __shared__ float qn[8 * 64];
    __shared__ float qs[8 * 64];
    __shared__ float logit[64 * 9];
    __shared__ float att[64 * 9];
    __shared__ float denom[8];
    __shared__ float red[16];
    __shared__ float updL[8 * 64];
    __shared__ float gxh[8 * 192];
    __shared__ float ghh2[8 * 192];
    __shared__ float sumv[64];

    const int b = blockIdx.x;
    const int t = threadIdx.x;
    const int s = t >> 6;
    const int d = t & 63;
    const size_t base = (size_t)b << 14;     // pixel base for this image
    const _Float16* keysB = keys + base * 64;
    const _Float16* valsB = vals + base * 64;

    // per-thread staging slot: 64 rows x 64 halfs, one b128 per thread
    const int rr  = t >> 3;
    const int seg = (t & 7) << 3;

    // load q
    qs[t] = qinit[(size_t)b * 512 + t];

    // sumvals[d] = sum_n vals[n][d]
    {
        float pv = 0.f;
        for (int n = s; n < HWSZ; n += 8)
            pv += (float)valsB[(size_t)n * 64 + d];
        gxh[t] = pv;
        __syncthreads();
        if (t < 64) {
            float a = 0.f;
            for (int j = 0; j < 8; j++) a += gxh[j * 64 + t];
            sumv[t] = a;
        }
        __syncthreads();
    }

    for (int iter = 0; iter < 3; iter++) {
        // ---- qn = LayerNorm(qs) ----
        if (t < 8) {
            float m = 0.f, m2 = 0.f;
            for (int j = 0; j < 64; j++) {
                float v = qs[t * 64 + j];
                m += v; m2 += v * v;
            }
            m *= (1.f / 64.f);
            float var = fmaxf(m2 * (1.f / 64.f) - m * m, 0.f);
            red[t * 2] = m;
            red[t * 2 + 1] = rsqrtf(var + 1e-5f);
        }
        __syncthreads();
        qn[t] = (qs[t] - red[s * 2]) * red[s * 2 + 1] * qg[d] + qb[d];
        if (t < 8) denom[t] = 0.f;
        __syncthreads();

        float accu = 0.f;                    // upd_un[s][d]
        for (int n0 = 0; n0 < HWSZ; n0 += 64) {
            int goff = (((n0 + rr) << 6) + seg) * 2;
            async_b128_to_lds(&kch[rr * 72 + seg], keysB, goff);
            async_b128_to_lds(&vch[rr * 72 + seg], valsB, goff);
            if (n0 + 64 < HWSZ) {            // prefetch next chunk into L2/L0
                __builtin_prefetch(&keysB[(size_t)(n0 + 64 + rr) * 64 + seg], 0, 1);
                __builtin_prefetch(&valsB[(size_t)(n0 + 64 + rr) * 64 + seg], 0, 1);
            }
            wait_async0();
            __syncthreads();
            // logits: thread t -> slot s, local n = d
            {
                float a = 0.f;
                for (int j = 0; j < 64; j++)
                    a += qn[s * 64 + j] * (float)kch[d * 72 + j];
                logit[d * 9 + s] = a * 0.125f;   // 1/sqrt(64)
            }
            __syncthreads();
            // softmax over slot axis
            if (t < 64) {
                float mx = -1e30f;
                for (int ss = 0; ss < 8; ss++) mx = fmaxf(mx, logit[t * 9 + ss]);
                float e[8], sm = 0.f;
                for (int ss = 0; ss < 8; ss++) { e[ss] = expf(logit[t * 9 + ss] - mx); sm += e[ss]; }
                float inv = 1.f / sm;
                for (int ss = 0; ss < 8; ss++) att[t * 9 + ss] = e[ss] * inv;
            }
            __syncthreads();
            if (t < 8) {
                float a = 0.f;
                for (int j = 0; j < 64; j++) a += att[j * 9 + t];
                denom[t] += a;
            }
            // accumulate unnormalized update
            for (int j = 0; j < 64; j++)
                accu += att[j * 9 + s] * (float)vch[j * 72 + d];
            __syncthreads();
        }
        updL[t] = accu / denom[s] + 1e-8f * sumv[d];
        __syncthreads();

        // ---- GRU gates: gx = upd @ wih^T + bih ; gh = h @ whh^T + bhh ----
        for (int rep = 0; rep < 3; rep++) {
            int idx = t + rep * 512;
            int ss  = idx / 192;
            int jj  = idx - ss * 192;
            float ax = bih[jj], ah = bhh[jj];
            for (int dd = 0; dd < 64; dd++) {
                ax += updL[ss * 64 + dd] * wih[jj * 64 + dd];
                ah += qs[ss * 64 + dd]   * whh[jj * 64 + dd];
            }
            gxh[ss * 192 + jj]  = ax;
            ghh2[ss * 192 + jj] = ah;
        }
        __syncthreads();
        {
            float xr = gxh[s * 192 + d],        hr = ghh2[s * 192 + d];
            float xz = gxh[s * 192 + 64 + d],   hz = ghh2[s * 192 + 64 + d];
            float xn = gxh[s * 192 + 128 + d],  hn = ghh2[s * 192 + 128 + d];
            float r  = 1.f / (1.f + expf(-(xr + hr)));
            float z  = 1.f / (1.f + expf(-(xz + hz)));
            float nn = tanhf(xn + r * hn);
            float nq = (1.f - z) * nn + z * qs[t];
            __syncthreads();
            qs[t] = nq;
        }
        __syncthreads();
    }

    // ---- final attn: write q, att, wts ----
    dout[(size_t)b * 512 + t] = qs[t];
    if (t < 8) {
        float m = 0.f, m2 = 0.f;
        for (int j = 0; j < 64; j++) { float v = qs[t * 64 + j]; m += v; m2 += v * v; }
        m *= (1.f / 64.f);
        float var = fmaxf(m2 * (1.f / 64.f) - m * m, 0.f);
        red[t * 2] = m;
        red[t * 2 + 1] = rsqrtf(var + 1e-5f);
    }
    __syncthreads();
    qn[t] = (qs[t] - red[s * 2]) * red[s * 2 + 1] * qg[d] + qb[d];
    if (t < 8) denom[t] = 0.f;
    __syncthreads();

    float* attO = dout + 32768 + ((size_t)b * 8) * HWSZ;
    float* wtsO = dout + 32768 + 8388608 + ((size_t)b * 8) * HWSZ;

    for (int n0 = 0; n0 < HWSZ; n0 += 64) {
        int goff = (((n0 + rr) << 6) + seg) * 2;
        async_b128_to_lds(&kch[rr * 72 + seg], keysB, goff);
        wait_async0();
        __syncthreads();
        {
            float a = 0.f;
            for (int j = 0; j < 64; j++)
                a += qn[s * 64 + j] * (float)kch[d * 72 + j];
            logit[d * 9 + s] = a * 0.125f;
        }
        __syncthreads();
        if (t < 64) {
            float mx = -1e30f;
            for (int ss = 0; ss < 8; ss++) mx = fmaxf(mx, logit[t * 9 + ss]);
            float e[8], sm = 0.f;
            for (int ss = 0; ss < 8; ss++) { e[ss] = expf(logit[t * 9 + ss] - mx); sm += e[ss]; }
            float inv = 1.f / sm;
            for (int ss = 0; ss < 8; ss++) att[t * 9 + ss] = e[ss] * inv;
        }
        __syncthreads();
        if (t < 8) {
            float a = 0.f;
            for (int j = 0; j < 64; j++) a += att[j * 9 + t];
            denom[t] += a;
        }
        attO[(size_t)s * HWSZ + n0 + d] = att[d * 9 + s];
        __syncthreads();
    }
    __threadfence();
    __syncthreads();
    for (int i = t; i < 8 * HWSZ; i += 512) {
        int ss = i >> 14;
        wtsO[i] = attO[i] / denom[ss] + 1e-8f;
    }
}

// ---------------------------------------------------------------------------
extern "C" void kernel_launch(void* const* d_in, const int* in_sizes, int n_in,
                              void* d_out, int out_size, void* d_ws, size_t ws_size,
                              hipStream_t stream) {
    const float* data    = (const float*)d_in[0];
    const float* c1w     = (const float*)d_in[1];
    const float* c1b     = (const float*)d_in[2];
    const float* c2w     = (const float*)d_in[3];
    const float* c2b     = (const float*)d_in[4];
    const float* c3w     = (const float*)d_in[5];
    const float* c3b     = (const float*)d_in[6];
    const float* pos_w   = (const float*)d_in[7];
    const float* pos_b   = (const float*)d_in[8];
    const float* dataN_g = (const float*)d_in[9];
    const float* dataN_b = (const float*)d_in[10];
    const float* m1w     = (const float*)d_in[11];
    const float* m1b     = (const float*)d_in[12];
    const float* m2w     = (const float*)d_in[13];
    const float* m2b     = (const float*)d_in[14];
    const float* qN_g    = (const float*)d_in[15];
    const float* qN_b    = (const float*)d_in[16];
    const float* kw      = (const float*)d_in[17];
    const float* kb      = (const float*)d_in[18];
    const float* vw      = (const float*)d_in[19];
    const float* vb      = (const float*)d_in[20];
    const float* wih     = (const float*)d_in[21];
    const float* whh     = (const float*)d_in[22];
    const float* bih     = (const float*)d_in[23];
    const float* bhh     = (const float*)d_in[24];
    const float* qinit   = (const float*)d_in[25];

    const size_t SZ = (size_t)NPIX * 64;     // halfs per big buffer
    _Float16* buf0 = (_Float16*)d_ws;
    _Float16* buf1 = buf0 + SZ;
    _Float16* w2T  = buf1 + SZ;
    _Float16* w3T  = w2T + 64 * KCONV;
    _Float16* m1T  = w3T + 64 * KCONV;
    _Float16* m2T  = m1T + 4096;
    _Float16* kT   = m2T + 4096;
    _Float16* vT   = kT + 4096;

    // 1) weight prep
    prep_weights_k<<<(64 * KCONV + 255) / 256, 256, 0, stream>>>(
        c2w, c3w, m1w, m2w, kw, vw, w2T, w3T, m1T, m2T, kT, vT);

    // 2) conv1 (direct) -> buf0 (f16 NCHW)
    conv1_k<<<NPIX / 256, 256, 0, stream>>>(data, c1w, c1b, buf0);

    // 3) conv2 (WMMA implicit GEMM, relu) -> buf1
    conv_wmma_k<<<NPIX / 16, 128, 0, stream>>>(buf0, w2T, c2b, buf1, 1);

    // 4) conv3 (WMMA, no relu) -> buf0
    conv_wmma_k<<<NPIX / 16, 128, 0, stream>>>(buf1, w3T, c3b, buf0, 0);

    // 5) pos-embed + LN -> buf1 (row-major [pixel][64] f16)
    lnpos_k<<<NPIX / 256, 256, 0, stream>>>(buf0, pos_w, pos_b, dataN_g, dataN_b, buf1);

    // 6) mlp1 (relu) -> buf0 ; 7) mlp2 -> buf1 (= enc)
    rowgemm_k<<<NPIX / 16, 128, 0, stream>>>(buf1, m1T, m1b, buf0, 1);
    rowgemm_k<<<NPIX / 16, 128, 0, stream>>>(buf0, m2T, m2b, buf1, 0);

    // 8) fused toK/toV: keys in-place over enc (buf1), vals -> buf0
    kv_k<<<NPIX / 16, 128, 0, stream>>>(buf1, kT, vT, kb, vb, buf1, buf0);

    // 9) slot attention iterations + outputs
    attn_k<<<BB, 512, 0, stream>>>(buf1, buf0, qinit, qN_g, qN_b,
                                   wih, whh, bih, bhh, (float*)d_out);
}
